// MIHashLoss_52398601011962
// MI455X (gfx1250) — matrix operand order repaired and checked by reading, not compiled, not run on hardware
//
#include <hip/hip_runtime.h>
#include <hip/hip_bf16.h>
#include <math.h>

typedef _Float16 v8h  __attribute__((ext_vector_type(8)));
typedef _Float16 v16h __attribute__((ext_vector_type(16)));
typedef float    v8f  __attribute__((ext_vector_type(8)));

#define NROWS   8192
#define NBIT    64
#define NCLASS  100
#define NB      16
#define NM1     8191.0f
#define EPSV    1e-7f

// workspace layout (bytes), all 256B aligned
#define PHI_OFF   0                         // 8192*64 f16  = 1,048,576
#define LAB_OFF   (NROWS * NBIT * 2)        // 8192 int     = 32,768
#define CNT_OFF   (LAB_OFF + NROWS * 4)     // 128 int
#define PDCP_OFF  (CNT_OFF + 512)           // 8192*16 f32
#define PDCN_OFF  (PDCP_OFF + NROWS * NB * 4)

static __device__ __forceinline__ void gatomic_add(float* p, float v) {
    __hip_atomic_fetch_add(p, v, __ATOMIC_RELAXED, __HIP_MEMORY_SCOPE_AGENT);
}
static __device__ __forceinline__ void satomic_add(float* p, float v) {
    __hip_atomic_fetch_add(p, v, __ATOMIC_RELAXED, __HIP_MEMORY_SCOPE_WORKGROUP);
}

// ---------------- init: zero histograms, class counts, output -------------
__global__ void mihash_init(float* pDCp, float* pDCn, int* cnt, float* out) {
    int idx = blockIdx.x * 256 + threadIdx.x;
    if (idx < NROWS * NB) { pDCp[idx] = 0.0f; pDCn[idx] = 0.0f; }
    if (idx < 128) cnt[idx] = 0;
    if (idx == 0) out[0] = 0.0f;
}

// ---------------- phi = tanh(u/2), stored f16 ------------------------------
__global__ void mihash_phi(const float* __restrict__ u, _Float16* __restrict__ phi) {
    int idx = blockIdx.x * 256 + threadIdx.x;
    if (idx < NROWS * NBIT) {
        phi[idx] = (_Float16)tanhf(0.5f * u[idx]);
    }
}

// ---------------- one-hot -> label, class counts ---------------------------
__global__ void mihash_labels(const float* __restrict__ y, int* __restrict__ lab,
                              int* __restrict__ cnt) {
    int i = blockIdx.x * 256 + threadIdx.x;
    if (i < NROWS) {
        const float* row = y + (size_t)i * NCLASS;
        int c = 0;
        for (int k = 0; k < NCLASS; ++k) {
            if (row[k] > 0.5f) { c = k; break; }
        }
        lab[i] = c;
        atomicAdd(&cnt[c], 1);
    }
}

// ---------------- fused WMMA GEMM + triangular-pulse histogram -------------
// grid = (8192/64, 8); block = 256 (8 waves; 4 row-waves x 2 col-waves)
// each block: 64 rows x 1024-column strip, inner loop 32 cols per step.
// Each col-wave gets a private 64x32 LDS histogram -> zero inter-wave
// contention (row-waves cover disjoint rows).
__launch_bounds__(256)
__global__ void mihash_hist(const _Float16* __restrict__ phi,
                            const int* __restrict__ lab,
                            float* __restrict__ pDCp,
                            float* __restrict__ pDCn) {
    __shared__ float hist[2 * 64 * 32];   // [wave_col][row][bin]; 0-15 = p, 16-31 = n

    const int tid = threadIdx.x;
    for (int idx = tid; idx < 2 * 64 * 32; idx += 256) hist[idx] = 0.0f;
    __syncthreads();

    const int wave = tid >> 5;
    const int lane = tid & 31;
    const int g    = lane >> 4;   // lane group 0/1
    const int n16  = lane & 15;
    const int wave_row = wave & 3;   // 4 row-waves  -> 64 rows
    const int wave_col = wave >> 2;  // 2 col-waves  -> 32 cols / iter

    const int i_block = blockIdx.x * 64;
    const int j_start = blockIdx.y * 1024;

    // ---- preload A fragments for K=0..63 (contiguous K per lane half, same order as B) ----
    // lane (g, M=lane%16): VGPR0-7 hold K = g*16+0..15
    const int mrow = i_block + wave_row * 16 + n16;
    const _Float16* arow = phi + (size_t)mrow * NBIT;
    v8h a00 = *(const v8h*)(arow + g * 16);
    v8h a01 = *(const v8h*)(arow + 8 + g * 16);
    v8h a10 = *(const v8h*)(arow + 32 + g * 16);
    v8h a11 = *(const v8h*)(arow + 40 + g * 16);
    v16h aLo, aHi;
    #pragma unroll
    for (int t = 0; t < 8; ++t) {
        aLo[t] = a00[t]; aLo[8 + t] = a01[t];
        aHi[t] = a10[t]; aHi[8 + t] = a11[t];
    }

    // labels for the 8 rows this lane's accumulator covers (M = r + 8*g)
    int lab_i[8];
    #pragma unroll
    for (int r = 0; r < 8; ++r)
        lab_i[r] = lab[i_block + wave_row * 16 + 8 * g + r];

    float* const hbase = &hist[wave_col * 2048 + (wave_row * 16 + 8 * g) * 32];

    for (int jt = 0; jt < 1024; jt += 32) {
        // ---- B fragment: lane (g, N) holds B[K = g*16+0..15][N] = phi[j][K..] ----
        const int jcol = j_start + jt + wave_col * 16 + n16;
        const _Float16* brow = phi + (size_t)jcol * NBIT;
        v16h bLo = *(const v16h*)(brow + g * 16);
        v16h bHi = *(const v16h*)(brow + 32 + g * 16);

        v8f c = {0.f, 0.f, 0.f, 0.f, 0.f, 0.f, 0.f, 0.f};
        c = __builtin_amdgcn_wmma_f32_16x16x32_f16(false, aLo, false, bLo,
                                                   (short)0, c, false, false);
        c = __builtin_amdgcn_wmma_f32_16x16x32_f16(false, aHi, false, bHi,
                                                   (short)0, c, false, false);

        const int lab_j = lab[jcol];

        #pragma unroll
        for (int r = 0; r < 8; ++r) {
            const int irow = i_block + wave_row * 16 + 8 * g + r;
            // t = dist/delta = (64 - c)*0.5*0.25 = 8 - c*0.125
            float t = fmaf(c[r], -0.125f, 8.0f);
            t = fminf(fmaxf(t, 0.0f), 15.9999f);
            const int   b0   = (int)t;
            const float frac = t - (float)b0;
            const bool  last = (b0 >= NB - 1);
            const int   b1   = last ? (NB - 1) : (b0 + 1);
            const float w1   = last ? 0.0f : frac;
            const bool same  = (lab_i[r] == lab_j) && (irow != jcol);
            float* base = hbase + r * 32 + (same ? 0 : 16);
            satomic_add(base + b0, 1.0f - frac);
            satomic_add(base + b1, w1);   // branchless: adds 0 when b0 == 15
        }
    }
    __syncthreads();

    // flush block-local histograms (merge the two col-wave copies) to global
    for (int idx = tid; idx < 64 * 32; idx += 256) {
        int lrow = idx >> 5, col = idx & 31;
        float v = hist[idx] + hist[2048 + idx];
        float* dst = (col < NB)
            ? (pDCp + (size_t)(i_block + lrow) * NB + col)
            : (pDCn + (size_t)(i_block + lrow) * NB + (col - NB));
        gatomic_add(dst, v);
    }
}

// ---------------- entropies + scalar reduction -----------------------------
__global__ void mihash_finalize(const float* __restrict__ pDCp,
                                const float* __restrict__ pDCn,
                                const int* __restrict__ lab,
                                const int* __restrict__ cnt,
                                float* __restrict__ out) {
    __shared__ float red[256];
    const int i = blockIdx.x * 256 + threadIdx.x;
    float term = 0.0f;
    if (i < NROWS) {
        float p[NB], n[NB], sp = 0.0f, sn = 0.0f;
        #pragma unroll
        for (int b = 0; b < NB; ++b) {
            p[b] = pDCp[(size_t)i * NB + b]; sp += p[b];
            n[b] = pDCn[(size_t)i * NB + b]; sn += n[b];
        }
        float prCp = (float)(cnt[lab[i]] - 1) / NM1;
        float prCn = 1.0f - prCp;
        float invp = (sp > 0.0f) ? 1.0f / sp : 1.0f;
        float invn = (sn > 0.0f) ? 1.0f / sn : 1.0f;
        float entD = 0.0f, entP = 0.0f, entN = 0.0f;
        #pragma unroll
        for (int b = 0; b < NB; ++b) {
            float pd = (p[b] + n[b]) * (1.0f / NM1);
            entD -= pd * logf(pd + EPSV);
            float ph = p[b] * invp;
            entP -= ph * logf(ph + EPSV);
            float nh = n[b] * invn;
            entN -= nh * logf(nh + EPSV);
        }
        term = entD - (prCp * entP + prCn * entN);
    }
    red[threadIdx.x] = term;
    __syncthreads();
    for (int s = 128; s > 0; s >>= 1) {
        if (threadIdx.x < s) red[threadIdx.x] += red[threadIdx.x + s];
        __syncthreads();
    }
    if (threadIdx.x == 0) gatomic_add(out, red[0]);
}

extern "C" void kernel_launch(void* const* d_in, const int* in_sizes, int n_in,
                              void* d_out, int out_size, void* d_ws, size_t ws_size,
                              hipStream_t stream) {
    (void)in_sizes; (void)n_in; (void)out_size; (void)ws_size;
    const float* u = (const float*)d_in[0];
    const float* y = (const float*)d_in[1];
    float* out = (float*)d_out;

    char* ws = (char*)d_ws;
    _Float16* phi = (_Float16*)(ws + PHI_OFF);
    int*  lab  = (int*)(ws + LAB_OFF);
    int*  cnt  = (int*)(ws + CNT_OFF);
    float* pDCp = (float*)(ws + PDCP_OFF);
    float* pDCn = (float*)(ws + PDCN_OFF);

    mihash_init<<<(NROWS * NB + 255) / 256, 256, 0, stream>>>(pDCp, pDCn, cnt, out);
    mihash_phi<<<(NROWS * NBIT + 255) / 256, 256, 0, stream>>>(u, phi);
    mihash_labels<<<(NROWS + 255) / 256, 256, 0, stream>>>(y, lab, cnt);

    dim3 grid(NROWS / 64, 8);   // 128 row-tiles x 8 column strips = 1024 blocks
    mihash_hist<<<grid, 256, 0, stream>>>(phi, lab, pDCp, pDCn);

    mihash_finalize<<<NROWS / 256, 256, 0, stream>>>(pDCp, pDCn, lab, cnt, out);
}
